// DynamicAggRecModel_24275155157398
// MI455X (gfx1250) — compile-verified, matrix operand order checked by reading, not run on hardware
//
#include <hip/hip_runtime.h>

typedef __attribute__((ext_vector_type(16))) _Float16 v16h;
typedef __attribute__((ext_vector_type(8)))  _Float16 v8h;
typedef __attribute__((ext_vector_type(4)))  _Float16 v4h;
typedef __attribute__((ext_vector_type(8)))  float    v8f;

#define BATCH 16384
#define HLEN  50
#define EMB   64
#define FEAT  64
#define ROWS  32          // users per block in kernel 1

__device__ __forceinline__ void cvt_store16(const float4 r[4], _Float16* __restrict__ dst) {
#pragma unroll
    for (int i = 0; i < 4; i++) {
        v4h h;
        h[0] = (_Float16)r[i].x; h[1] = (_Float16)r[i].y;
        h[2] = (_Float16)r[i].z; h[3] = (_Float16)r[i].w;
        *(v4h*)(dst + i * 4) = h;
    }
}

__device__ __forceinline__ v16h load_a_frag(const _Float16* __restrict__ row, int koff) {
    v8h lo = *(const v8h*)(row + koff);
    v8h hi = *(const v8h*)(row + koff + 16);
    v16h a;
#pragma unroll
    for (int t = 0; t < 8; t++) { a[t] = lo[t]; a[t + 8] = hi[t]; }
    return a;
}

// ---------------------------------------------------------------------------
// Kernel 1: item reps (history + target) and weighted pooling.
//   32 users per block, 8 waves = 2 M-tiles x 4 N-tiles, double-buffered X.
// ---------------------------------------------------------------------------
__global__ __launch_bounds__(256) void rep_pool_kernel(
    const int*   __restrict__ hist_idx,   // (B,H)
    const float* __restrict__ hist_feat,  // (B,H,64)
    const float* __restrict__ hist_rat,   // (B,H)
    const int*   __restrict__ tgt_idx,    // (B,)
    const float* __restrict__ tgt_feat,   // (B,64)
    const float* __restrict__ emb,        // (NUM_ITEMS,64)
    const float* __restrict__ fw,         // (128,64)
    const float* __restrict__ fb,         // (64,)
    float* __restrict__ user_vec,         // (B,64) ws
    float* __restrict__ tgt_rep)          // (B,64) ws
{
    __shared__ _Float16 sW[128 * 64];              // fusion_w f16, K x N
    __shared__ _Float16 sX[2][ROWS][136];          // double-buffered X tiles
    __shared__ int      sIdx[HLEN + 1][ROWS];      // all gather indices
    __shared__ float    sPoolT[HLEN][ROWS];        // (rating-3), transposed
    __shared__ float    sAbsW[ROWS];

    const int tid  = threadIdx.x;
    const int lane = tid & 31;
    const int wid  = tid >> 5;
    const int b0   = blockIdx.x * ROWS;

    // ---- stage fusion_w -> f16, indices, pooling weights ----
    {
        const float4* w4 = (const float4*)fw;
#pragma unroll 2
        for (int i = tid; i < (128 * 64) / 4; i += 256) {
            float4 v = w4[i];
            v4h h;
            h[0] = (_Float16)v.x; h[1] = (_Float16)v.y;
            h[2] = (_Float16)v.z; h[3] = (_Float16)v.w;
            *(v4h*)&sW[i * 4] = h;
        }
        for (int e = tid; e < ROWS * HLEN; e += 256) {
            const int r = e / HLEN, s = e % HLEN;
            sIdx[s][r]   = hist_idx[(b0 + r) * HLEN + s];
            sPoolT[s][r] = hist_rat[(b0 + r) * HLEN + s] - 3.0f;
        }
        if (tid < ROWS) sIdx[HLEN][tid] = tgt_idx[b0 + tid];
    }
    __syncthreads();

    if (tid < ROWS) {
        float a = 0.f;
#pragma unroll 5
        for (int s = 0; s < HLEN; s++) a += fabsf(sPoolT[s][tid]);
        sAbsW[tid] = a;
    }

    const int mtile = wid >> 2;                    // 0 or 1 (M block of 16)
    const int ntile = wid & 3;                     // 0..3 (N block of 16)
    const int ncol  = (ntile << 4) + (lane & 15);
    const int khalf = (lane >> 4) << 3;
    const int arow  = (mtile << 4) + (lane & 15);
    const int mrow0 = (mtile << 4) + ((lane >> 4) << 3);   // first C row this lane owns

    // B fragments of fusion_w (constant across all steps)
    v16h bfr[4];
#pragma unroll
    for (int kc = 0; kc < 4; kc++) {
        const int kb = kc * 32 + khalf;
#pragma unroll
        for (int t = 0; t < 8; t++) {
            bfr[kc][t]     = sW[(kb + t) * 64 + ncol];
            bfr[kc][t + 8] = sW[(kb + 16 + t) * 64 + ncol];
        }
    }
    const float bias = fb[ncol];

    const int xrow = tid >> 3;   // 32 rows, 8 threads per row
    const int q    = tid & 7;

    // loader lambda: fetch step `s` X chunk into registers (no LDS yet)
    auto load_step = [&](float4 r[4], int s) {
        const float* src;
        if (q < 4) {
            const int gidx = sIdx[s][xrow];
            src = emb + (size_t)gidx * EMB + q * 16;
        } else if (s < HLEN) {
            src = hist_feat + ((size_t)(b0 + xrow) * HLEN + s) * FEAT + (q - 4) * 16;
        } else {
            src = tgt_feat + (size_t)(b0 + xrow) * FEAT + (q - 4) * 16;
        }
        const float4* s4 = (const float4*)src;
#pragma unroll
        for (int i = 0; i < 4; i++) r[i] = s4[i];
    };

    float acc[8] = {0.f, 0.f, 0.f, 0.f, 0.f, 0.f, 0.f, 0.f};

    // preamble: fill buffer 0
    float4 rg[4];
    load_step(rg, 0);
    cvt_store16(rg, &sX[0][xrow][q * 16]);
    __syncthreads();

    for (int s = 0; s <= HLEN; s++) {
        const int buf = s & 1;

        // issue next step's global loads (hidden behind the WMMA chain)
        if (s < HLEN) load_step(rg, s + 1);

        // batch all A-fragment LDS loads, then the WMMA chain
        v16h afr[4];
#pragma unroll
        for (int kc = 0; kc < 4; kc++)
            afr[kc] = load_a_frag(&sX[buf][arow][0], kc * 32 + khalf);
        v8f c = {};
#pragma unroll
        for (int kc = 0; kc < 4; kc++)
            c = __builtin_amdgcn_wmma_f32_16x16x32_f16(
                    false, afr[kc], false, bfr[kc], (short)0, c, false, false);

        if (s < HLEN) {
#pragma unroll
            for (int j = 0; j < 8; j++) {
                float v = c[j] + bias;
                v = v > 0.f ? v : 0.f;
                acc[j] += v * sPoolT[s][mrow0 + j];
            }
            // convert + store next-step tile into the other buffer
            cvt_store16(rg, &sX[buf ^ 1][xrow][q * 16]);
        } else {
#pragma unroll
            for (int j = 0; j < 8; j++) {
                float v = c[j] + bias;
                v = v > 0.f ? v : 0.f;
                tgt_rep[(size_t)(b0 + mrow0 + j) * EMB + ncol] = v;
            }
        }
        __syncthreads();
    }

#pragma unroll
    for (int j = 0; j < 8; j++) {
        const int row = mrow0 + j;
        user_vec[(size_t)(b0 + row) * EMB + ncol] = acc[j] / (sAbsW[row] + 1e-8f);
    }
}

// ---------------------------------------------------------------------------
// Kernel 2: rating MLP.  x=(user_vec|tgt_rep) -> relu@w1 -> relu@w2 -> @w3.
// ---------------------------------------------------------------------------
__global__ __launch_bounds__(128) void mlp_kernel(
    const float* __restrict__ user_vec,   // (B,64)
    const float* __restrict__ tgt_rep,    // (B,64)
    const float* __restrict__ w1, const float* __restrict__ b1,   // (128,64),(64,)
    const float* __restrict__ w2, const float* __restrict__ b2,   // (64,32),(32,)
    const float* __restrict__ w3, const float* __restrict__ b3,   // (32,1),(1,)
    float* __restrict__ out)              // (B,)
{
    __shared__ _Float16 sW1[128 * 64];
    __shared__ _Float16 sW2[64 * 32];
    __shared__ _Float16 sX[16][136];
    __shared__ _Float16 sH1[16][80];
    __shared__ float    sH2[16][33];

    const int tid  = threadIdx.x;
    const int lane = tid & 31;
    const int wid  = tid >> 5;
    const int b0   = blockIdx.x * 16;

    {
        const float4* w4 = (const float4*)w1;
#pragma unroll 4
        for (int i = tid; i < (128 * 64) / 4; i += 128) {
            float4 v = w4[i];
            v4h h;
            h[0] = (_Float16)v.x; h[1] = (_Float16)v.y;
            h[2] = (_Float16)v.z; h[3] = (_Float16)v.w;
            *(v4h*)&sW1[i * 4] = h;
        }
        const float4* v4 = (const float4*)w2;
#pragma unroll
        for (int i = tid; i < (64 * 32) / 4; i += 128) {
            float4 v = v4[i];
            v4h h;
            h[0] = (_Float16)v.x; h[1] = (_Float16)v.y;
            h[2] = (_Float16)v.z; h[3] = (_Float16)v.w;
            *(v4h*)&sW2[i * 4] = h;
        }
    }
    {
        const int xrow = tid >> 3;
        const int q    = tid & 7;
        const float* src = (q < 4)
            ? user_vec + (size_t)(b0 + xrow) * 64 + q * 16
            : tgt_rep  + (size_t)(b0 + xrow) * 64 + (q - 4) * 16;
        const float4* s4 = (const float4*)src;
        float4 r[4];
#pragma unroll
        for (int i = 0; i < 4; i++) r[i] = s4[i];
        cvt_store16(r, &sX[xrow][q * 16]);
    }
    __syncthreads();

    const int ncol  = (wid << 4) + (lane & 15);
    const int khalf = (lane >> 4) << 3;
    const int arow  = lane & 15;
    const int r8    = (lane >> 4) << 3;

    // ---- layer 1: h1 = relu(x @ w1 + b1), 16x64 split over 4 waves ----
    {
        v16h bfr[4];
#pragma unroll
        for (int kc = 0; kc < 4; kc++) {
            const int kb = kc * 32 + khalf;
#pragma unroll
            for (int t = 0; t < 8; t++) {
                bfr[kc][t]     = sW1[(kb + t) * 64 + ncol];
                bfr[kc][t + 8] = sW1[(kb + 16 + t) * 64 + ncol];
            }
        }
        const float bias1 = b1[ncol];
        v16h afr[4];
#pragma unroll
        for (int kc = 0; kc < 4; kc++)
            afr[kc] = load_a_frag(&sX[arow][0], kc * 32 + khalf);
        v8f c = {};
#pragma unroll
        for (int kc = 0; kc < 4; kc++)
            c = __builtin_amdgcn_wmma_f32_16x16x32_f16(
                    false, afr[kc], false, bfr[kc], (short)0, c, false, false);
#pragma unroll
        for (int j = 0; j < 8; j++) {
            float v = c[j] + bias1;
            v = v > 0.f ? v : 0.f;
            sH1[j + r8][ncol] = (_Float16)v;
        }
    }
    __syncthreads();

    // ---- layer 2: h2 = relu(h1 @ w2 + b2), 16x32 on waves 0-1 ----
    if (wid < 2) {
        const int n2 = (wid << 4) + (lane & 15);
        v16h bf2[2];
#pragma unroll
        for (int kc = 0; kc < 2; kc++) {
            const int kb = kc * 32 + khalf;
#pragma unroll
            for (int t = 0; t < 8; t++) {
                bf2[kc][t]     = sW2[(kb + t) * 32 + n2];
                bf2[kc][t + 8] = sW2[(kb + 16 + t) * 32 + n2];
            }
        }
        const float bias2 = b2[n2];
        v16h af2[2];
#pragma unroll
        for (int kc = 0; kc < 2; kc++)
            af2[kc] = load_a_frag(&sH1[arow][0], kc * 32 + khalf);
        v8f c2 = {};
#pragma unroll
        for (int kc = 0; kc < 2; kc++)
            c2 = __builtin_amdgcn_wmma_f32_16x16x32_f16(
                     false, af2[kc], false, bf2[kc], (short)0, c2, false, false);
#pragma unroll
        for (int j = 0; j < 8; j++) {
            float v = c2[j] + bias2;
            sH2[j + r8][n2] = v > 0.f ? v : 0.f;
        }
    }
    __syncthreads();

    // ---- layer 3: pred = h2 @ w3 + b3 (32 -> 1), lanes 0..15 of wave 0 ----
    if (tid < 16) {
        float s = b3[0];
#pragma unroll 8
        for (int c = 0; c < 32; c++) s += sH2[tid][c] * w3[c];
        out[b0 + tid] = s;
    }
}

extern "C" void kernel_launch(void* const* d_in, const int* in_sizes, int n_in,
                              void* d_out, int out_size, void* d_ws, size_t ws_size,
                              hipStream_t stream) {
    const int*   hist_idx  = (const int*)  d_in[0];
    const float* hist_feat = (const float*)d_in[1];
    const float* hist_rat  = (const float*)d_in[2];
    const int*   tgt_idx   = (const int*)  d_in[3];
    const float* tgt_feat  = (const float*)d_in[4];
    const float* emb       = (const float*)d_in[5];
    const float* fw        = (const float*)d_in[6];
    const float* fb        = (const float*)d_in[7];
    const float* w1        = (const float*)d_in[8];
    const float* b1        = (const float*)d_in[9];
    const float* w2        = (const float*)d_in[10];
    const float* b2        = (const float*)d_in[11];
    const float* w3        = (const float*)d_in[12];
    const float* b3        = (const float*)d_in[13];

    float* user_vec = (float*)d_ws;                        // B*64 f32
    float* tgt_rep  = (float*)d_ws + (size_t)BATCH * EMB;  // B*64 f32

    rep_pool_kernel<<<dim3(BATCH / ROWS), dim3(256), 0, stream>>>(
        hist_idx, hist_feat, hist_rat, tgt_idx, tgt_feat,
        emb, fw, fb, user_vec, tgt_rep);
    mlp_kernel<<<dim3(BATCH / 16), dim3(128), 0, stream>>>(
        user_vec, tgt_rep, w1, b1, w2, b2, w3, b3, (float*)d_out);
}
